// GNN_44693429682362
// MI455X (gfx1250) — compile-verified, matrix operand order checked by reading, not run on hardware
//
#include <hip/hip_runtime.h>

// ---------------------------------------------------------------------------
// CDNA5 (gfx1250) GNN forward pass.
// All dense 128-col layers run through a bf16 WMMA GEMM kernel
// (v_wmma_f32_16x16x32_bf16), with concat-inputs implemented as gathered
// K-segments and bias/ELU/LayerNorm/residual fused in the epilogue.
// Intermediates are stored bf16 to halve HBM traffic (23.3 TB/s roofline).
// ---------------------------------------------------------------------------

typedef __attribute__((ext_vector_type(16))) __bf16 v16bf;
typedef __attribute__((ext_vector_type(8)))  __bf16 v8bf;
typedef __attribute__((ext_vector_type(8)))  float  v8f;
typedef __attribute__((ext_vector_type(2)))  __bf16 bf16x2;

struct ABpair { v8bf lo, hi; };   // 32B, bit-cast to v16bf fragment

__device__ __forceinline__ float bf2f(__bf16 b) {
    unsigned short s = __builtin_bit_cast(unsigned short, b);
    unsigned u = ((unsigned)s) << 16;
    return __builtin_bit_cast(float, u);
}
__device__ __forceinline__ __bf16 f2bf(float f) {
    unsigned u = __builtin_bit_cast(unsigned, f);
    unsigned r = (u + 0x7fffu + ((u >> 16) & 1u)) >> 16;
    return __builtin_bit_cast(__bf16, (unsigned short)r);
}
__device__ __forceinline__ bf16x2 f2bf2(float a, float b) {
    bf16x2 r; r[0] = f2bf(a); r[1] = f2bf(b); return r;
}
__device__ __forceinline__ float eluf(float x) {
    return x > 0.f ? x : (__expf(x) - 1.f);
}

#define TILE_M    128
#define LDS_PITCH 136   // bf16 pitch for transposed W chunk; avoids bank conflicts

// ---------------------------------------------------------------------------
// Generic fused MLP layer: Y[M,128] = epi( concat_segs(X)[M, NSEG*128] @ W + b )
//   NSEG segments of width 128; each segment has its own base pointer and an
//   optional row-gather index (for h[send] / h[recv] style inputs).
//   epi: ELU (hidden) or LayerNorm (+ optional residual) for final layers.
// ---------------------------------------------------------------------------
template <int NSEG, bool ELU, bool LN, bool RES>
__global__ __launch_bounds__(256) void mlp_wmma(
    const __bf16* __restrict__ s0, const int* __restrict__ i0,
    const __bf16* __restrict__ s1, const int* __restrict__ i1,
    const __bf16* __restrict__ s2, const int* __restrict__ i2,
    const float* __restrict__ W,     // [NSEG*128, 128] row-major f32
    const float* __restrict__ bias,  // [128]
    const float* __restrict__ ln_g, const float* __restrict__ ln_b,
    const __bf16* __restrict__ res,  // residual [M,128] or null
    __bf16* __restrict__ out, int M)
{
    __shared__ __bf16 ldsW[128 * LDS_PITCH];   // transposed chunk: [n][k]

    const int tid  = threadIdx.x;
    const int lane = tid & 31;
    const int wave = tid >> 5;
    const int ln15 = lane & 15;
    const int half = (lane >> 4) & 1;
    const int rowA = blockIdx.x * TILE_M + wave * 16 + ln15;  // A-operand row
    // Clamp instead of predicating loads: out-of-range rows compute garbage
    // that is discarded at the (guarded) store. Keeps all loads branch-free.
    const int rA = (rowA < M) ? rowA : (M - 1);

    v8f acc[8];
#pragma unroll
    for (int nb = 0; nb < 8; ++nb) {
        v8f z = {0.f, 0.f, 0.f, 0.f, 0.f, 0.f, 0.f, 0.f};
        acc[nb] = z;
    }

#pragma unroll
    for (int kc = 0; kc < NSEG; ++kc) {
        const __bf16* sp = (kc == 0) ? s0 : (kc == 1) ? s1 : s2;
        const int*    ip = (kc == 0) ? i0 : (kc == 1) ? i1 : i2;

        // ---- stage W chunk (128k x 128n) into LDS, bf16, transposed ----
        __syncthreads();
        const float* Wc = W + (size_t)kc * 128 * 128;
#pragma unroll 4
        for (int j = 0; j < 32; ++j) {
            int idx = j * 256 + tid;             // 0..8191
            int k = (idx >> 7) << 1, n = idx & 127;
            bf16x2 pk = f2bf2(Wc[k * 128 + n], Wc[(k + 1) * 128 + n]);
            *(bf16x2*)(ldsW + n * LDS_PITCH + k) = pk;   // 4B-aligned b32 store
        }

        // ---- preload this lane's A-row segment: 8 x b128 (overlaps barrier) ----
        long long gr = ip ? (long long)ip[rA] : (long long)rA;
        const v8bf* arow = (const v8bf*)(sp + gr * 128 + half * 8);  // 16B aligned
        v8bf ach[8];
#pragma unroll
        for (int c = 0; c < 8; ++c) ach[c] = arow[2 * c];  // k-offset = c*16

        __syncthreads();

#pragma unroll
        for (int kk = 0; kk < 4; ++kk) {
            // A fragment (ISA 16-bit 16x32 layout): halves at k = kk*32+half*8
            // (+0..7) and (+16..23) -> chunks 2*kk and 2*kk+1.
            ABpair ap; ap.lo = ach[2 * kk]; ap.hi = ach[2 * kk + 1];
            v16bf a = __builtin_bit_cast(v16bf, ap);
            // ---- 8 B fragments from LDS + WMMA accumulate ----
#pragma unroll
            for (int nb = 0; nb < 8; ++nb) {
                int n = nb * 16 + ln15;
                const __bf16* brow = ldsW + n * LDS_PITCH + kk * 32 + half * 8;
                ABpair bp;
                bp.lo = *(const v8bf*)(brow);        // ds_load_b128
                bp.hi = *(const v8bf*)(brow + 16);   // ds_load_b128
                v16bf b = __builtin_bit_cast(v16bf, bp);
                acc[nb] = __builtin_amdgcn_wmma_f32_16x16x32_bf16(
                    false, a, false, b, (short)0, acc[nb], false, false);
            }
        }
    }

    // ------------------------------ epilogue -------------------------------
    float bn[8], gn[8], lbn[8];
#pragma unroll
    for (int nb = 0; nb < 8; ++nb) {
        int n = nb * 16 + ln15;
        bn[nb] = bias[n];
        if (LN) { gn[nb] = ln_g[n]; lbn[nb] = ln_b[n]; }
    }
#pragma unroll
    for (int nb = 0; nb < 8; ++nb)
#pragma unroll
        for (int v = 0; v < 8; ++v)
            acc[nb][v] = acc[nb][v] + bn[nb];

    if (LN) {
        // D layout: vgpr v, lane-half h -> row v + 8*h; the 128 channels of a
        // row live in the 16-lane half-group x 8 N-tiles. Reduce nb locally,
        // then 16-lane butterfly (xor 8,4,2,1 stays within each half-group).
#pragma unroll
        for (int v = 0; v < 8; ++v) {
            float s = 0.f, q = 0.f;
#pragma unroll
            for (int nb = 0; nb < 8; ++nb) {
                float xx = acc[nb][v];
                s += xx; q += xx * xx;
            }
#pragma unroll
            for (int off = 8; off > 0; off >>= 1) {
                s += __shfl_xor(s, off, 32);
                q += __shfl_xor(q, off, 32);
            }
            float mu  = s * (1.f / 128.f);
            float var = q * (1.f / 128.f) - mu * mu;
            float inv = rsqrtf(var + 1e-5f);
#pragma unroll
            for (int nb = 0; nb < 8; ++nb)
                acc[nb][v] = (acc[nb][v] - mu) * inv * gn[nb] + lbn[nb];
        }
    }

#pragma unroll
    for (int v = 0; v < 8; ++v) {
        int rD = blockIdx.x * TILE_M + wave * 16 + v + 8 * half;
        if (rD < M) {
#pragma unroll
            for (int nb = 0; nb < 8; ++nb) {
                int n = nb * 16 + ln15;
                float xv = acc[nb][v];
                if (ELU) xv = eluf(xv);
                if (RES) xv += bf2f(res[(size_t)rD * 128 + n]);
                out[(size_t)rD * 128 + n] = f2bf(xv);
            }
        }
    }
}

// ---------------------------------------------------------------------------
// Tiny-K kernels (7->128, 3->128, 128->3) and scatter-add — negligible FLOPs.
// ---------------------------------------------------------------------------
__global__ void edge_feat_enc0(const float* __restrict__ x,
                               const float* __restrict__ pos,
                               const int* __restrict__ ei,
                               const float* __restrict__ W,  // [7,128]
                               const float* __restrict__ b,
                               __bf16* __restrict__ out, int E)
{
    __shared__ float f[7];
    int e = blockIdx.x;
    if (e >= E) return;
    int c = threadIdx.x;
    if (c == 0) {
        int s = ei[e], r = ei[E + e];
        float d0 = pos[s * 3 + 0] - pos[r * 3 + 0];
        float d1 = pos[s * 3 + 1] - pos[r * 3 + 1];
        float d2 = pos[s * 3 + 2] - pos[r * 3 + 2];
        f[0] = d0; f[1] = d1; f[2] = d2;
        f[3] = sqrtf(d0 * d0 + d1 * d1 + d2 * d2);
        f[4] = x[s * 3 + 0] - x[r * 3 + 0];
        f[5] = x[s * 3 + 1] - x[r * 3 + 1];
        f[6] = x[s * 3 + 2] - x[r * 3 + 2];
    }
    __syncthreads();
    float acc = b[c];
#pragma unroll
    for (int k = 0; k < 7; ++k) acc += f[k] * W[k * 128 + c];
    out[(size_t)e * 128 + c] = f2bf(eluf(acc));
}

__global__ void node_feat_enc0(const float* __restrict__ x,
                               const float* __restrict__ W,  // [3,128]
                               const float* __restrict__ b,
                               __bf16* __restrict__ out, int N)
{
    __shared__ float f[3];
    int n = blockIdx.x;
    if (n >= N) return;
    int c = threadIdx.x;
    if (c < 3) f[c] = x[n * 3 + c];
    __syncthreads();
    float acc = b[c];
#pragma unroll
    for (int k = 0; k < 3; ++k) acc += f[k] * W[k * 128 + c];
    out[(size_t)n * 128 + c] = f2bf(eluf(acc));
}

__global__ void scatter_add_edges(const __bf16* __restrict__ eb,
                                  const int* __restrict__ recv,
                                  float* __restrict__ agg, long long total)
{
    long long i = (long long)blockIdx.x * blockDim.x + threadIdx.x;
    if (i >= total) return;
    int e = (int)(i >> 7), c = (int)(i & 127);
    atomicAdd(&agg[(size_t)recv[e] * 128 + c], bf2f(eb[i]));
}

__global__ void cvt_f32_bf16(const float* __restrict__ in,
                             __bf16* __restrict__ out, long long n)
{
    long long i = (long long)blockIdx.x * blockDim.x + threadIdx.x;
    if (i < n) out[i] = f2bf(in[i]);
}

__global__ void decode_last(const __bf16* __restrict__ h,
                            const float* __restrict__ W,  // [128,3]
                            const float* __restrict__ b,
                            float* __restrict__ out, int N)
{
    int i = blockIdx.x * blockDim.x + threadIdx.x;
    if (i >= N * 3) return;
    int n = i / 3, c = i % 3;
    float acc = b[c];
    for (int k = 0; k < 128; ++k)
        acc += bf2f(h[(size_t)n * 128 + k]) * W[k * 3 + c];
    out[i] = acc;
}

// ---------------------------------------------------------------------------
// Host side: param indexing assumes recursive insertion-order flattening of
// setup_inputs(): x, edge_index, pos, then for each MLP: W0..W3, b0..b3
// (+ ln_g, ln_b when present); order: node_encoder, edge_encoder,
// node_decoder, mp0.edge, mp0.node, mp1.edge, mp1.node.
// ---------------------------------------------------------------------------
extern "C" void kernel_launch(void* const* d_in, const int* in_sizes, int n_in,
                              void* d_out, int out_size, void* d_ws, size_t ws_size,
                              hipStream_t stream)
{
    (void)n_in; (void)out_size; (void)ws_size;
    const float* x   = (const float*)d_in[0];
    const int*   ei  = (const int*)d_in[1];
    const float* pos = (const float*)d_in[2];
    const int N = in_sizes[0] / 3;
    const int E = in_sizes[1] / 2;

    auto P = [&](int i) { return (const float*)d_in[i]; };

    // workspace carve-up (256B aligned)
    size_t off = 0;
    auto carve = [&](size_t bytes) -> void* {
        void* p = (char*)d_ws + off;
        off += (bytes + 255) & ~(size_t)255;
        return p;
    };
    __bf16* e_buf = (__bf16*)carve((size_t)E * 128 * sizeof(__bf16));
    __bf16* t0    = (__bf16*)carve((size_t)E * 128 * sizeof(__bf16));
    __bf16* t1    = (__bf16*)carve((size_t)E * 128 * sizeof(__bf16));
    __bf16* h     = (__bf16*)carve((size_t)N * 128 * sizeof(__bf16));
    __bf16* u0    = (__bf16*)carve((size_t)N * 128 * sizeof(__bf16));
    __bf16* u1    = (__bf16*)carve((size_t)N * 128 * sizeof(__bf16));
    float*  aggf  = (float*) carve((size_t)N * 128 * sizeof(float));
    __bf16* aggh  = (__bf16*)carve((size_t)N * 128 * sizeof(__bf16));

    const int* send = ei;
    const int* recv = ei + E;
    dim3 blk(256);
    auto g = [](int M) { return dim3((unsigned)((M + TILE_M - 1) / TILE_M)); };

    const __bf16* nb16 = nullptr;
    const int*    ni   = nullptr;
    const float*  nf   = nullptr;

    // ---------------- node encoder: 3 -> 128 (ELU x3, final LN) ------------
    node_feat_enc0<<<dim3((unsigned)N), dim3(128), 0, stream>>>(x, P(3), P(7), u0, N);
    mlp_wmma<1, true,  false, false><<<g(N), blk, 0, stream>>>(u0, ni, nb16, ni, nb16, ni, P(4), P(8),  nf, nf, nb16, u1, N);
    mlp_wmma<1, true,  false, false><<<g(N), blk, 0, stream>>>(u1, ni, nb16, ni, nb16, ni, P(5), P(9),  nf, nf, nb16, u0, N);
    mlp_wmma<1, false, true,  false><<<g(N), blk, 0, stream>>>(u0, ni, nb16, ni, nb16, ni, P(6), P(10), P(11), P(12), nb16, h, N);

    // ---------------- edge encoder: 7 -> 128 -------------------------------
    edge_feat_enc0<<<dim3((unsigned)E), dim3(128), 0, stream>>>(x, pos, ei, P(13), P(17), t0, E);
    mlp_wmma<1, true,  false, false><<<g(E), blk, 0, stream>>>(t0, ni, nb16, ni, nb16, ni, P(14), P(18), nf, nf, nb16, t1, E);
    mlp_wmma<1, true,  false, false><<<g(E), blk, 0, stream>>>(t1, ni, nb16, ni, nb16, ni, P(15), P(19), nf, nf, nb16, t0, E);
    mlp_wmma<1, false, true,  false><<<g(E), blk, 0, stream>>>(t0, ni, nb16, ni, nb16, ni, P(16), P(20), P(21), P(22), nb16, e_buf, E);

    // ---------------- message passing x2 -----------------------------------
    for (int i = 0; i < 2; ++i) {
        int eb = 31 + 20 * i;   // edge-MLP params
        int nb = 41 + 20 * i;   // node-MLP params

        // edge MLP: concat(h[send], h[recv], e) @ W  (K = 384, fused gather)
        mlp_wmma<3, true,  false, false><<<g(E), blk, 0, stream>>>(h, send, h, recv, e_buf, ni, P(eb),     P(eb + 4), nf, nf, nb16, t0, E);
        mlp_wmma<1, true,  false, false><<<g(E), blk, 0, stream>>>(t0, ni, nb16, ni, nb16, ni, P(eb + 1), P(eb + 5), nf, nf, nb16, t1, E);
        mlp_wmma<1, true,  false, false><<<g(E), blk, 0, stream>>>(t1, ni, nb16, ni, nb16, ni, P(eb + 2), P(eb + 6), nf, nf, nb16, t0, E);
        // final: e = e + LN(mlp_out)   (in-place residual is element-local -> safe)
        mlp_wmma<1, false, true,  true ><<<g(E), blk, 0, stream>>>(t0, ni, nb16, ni, nb16, ni, P(eb + 3), P(eb + 7), P(eb + 8), P(eb + 9), e_buf, e_buf, E);

        // agg = segment_sum(e, recv)
        hipMemsetAsync(aggf, 0, (size_t)N * 128 * sizeof(float), stream);
        long long totE = (long long)E * 128;
        scatter_add_edges<<<dim3((unsigned)((totE + 255) / 256)), blk, 0, stream>>>(e_buf, recv, aggf, totE);
        long long totN = (long long)N * 128;
        cvt_f32_bf16<<<dim3((unsigned)((totN + 255) / 256)), blk, 0, stream>>>(aggf, aggh, totN);

        // node MLP: concat(h, agg) @ W  (K = 256)
        mlp_wmma<2, true,  false, false><<<g(N), blk, 0, stream>>>(h, ni, aggh, ni, nb16, ni, P(nb),     P(nb + 4), nf, nf, nb16, u0, N);
        mlp_wmma<1, true,  false, false><<<g(N), blk, 0, stream>>>(u0, ni, nb16, ni, nb16, ni, P(nb + 1), P(nb + 5), nf, nf, nb16, u1, N);
        mlp_wmma<1, true,  false, false><<<g(N), blk, 0, stream>>>(u1, ni, nb16, ni, nb16, ni, P(nb + 2), P(nb + 6), nf, nf, nb16, u0, N);
        // h = h + LN(mlp_out)
        mlp_wmma<1, false, true,  true ><<<g(N), blk, 0, stream>>>(u0, ni, nb16, ni, nb16, ni, P(nb + 3), P(nb + 7), P(nb + 8), P(nb + 9), h, h, N);
    }

    // ---------------- decoder: 128 -> 3 ------------------------------------
    mlp_wmma<1, true, false, false><<<g(N), blk, 0, stream>>>(h,  ni, nb16, ni, nb16, ni, P(23), P(27), nf, nf, nb16, u0, N);
    mlp_wmma<1, true, false, false><<<g(N), blk, 0, stream>>>(u0, ni, nb16, ni, nb16, ni, P(24), P(28), nf, nf, nb16, u1, N);
    mlp_wmma<1, true, false, false><<<g(N), blk, 0, stream>>>(u1, ni, nb16, ni, nb16, ni, P(25), P(29), nf, nf, nb16, u0, N);
    decode_last<<<dim3((unsigned)((N * 3 + 255) / 256)), blk, 0, stream>>>(u0, P(26), P(30), (float*)d_out, N);
}
